// MultiHeadedAttention_31628139167815
// MI455X (gfx1250) — matrix operand order repaired and checked by reading, not compiled, not run on hardware
//
#include <hip/hip_runtime.h>

#define DEV static __device__ __forceinline__

typedef __attribute__((ext_vector_type(16))) __bf16          v16bf;
typedef __attribute__((ext_vector_type(8)))  float           v8f;
typedef __attribute__((ext_vector_type(16))) unsigned short  v16u;
typedef __attribute__((ext_vector_type(8)))  unsigned short  v8u;
typedef __attribute__((ext_vector_type(4)))  unsigned short  v4u;
typedef __attribute__((ext_vector_type(4)))  unsigned int    u32x4;
typedef __attribute__((ext_vector_type(8)))  unsigned int    u32x8;
typedef __attribute__((ext_vector_type(4)))  float           vf4;   // native vectors for
typedef __attribute__((ext_vector_type(4)))  int             vi4;   // nontemporal builtins

constexpr int BATCH = 2;
constexpr int SEQ   = 2048;
constexpr int DM    = 1024;
constexpr int NH    = 16;
constexpr int DKH   = 64;            // per-head dim
constexpr int ROWS  = BATCH * SEQ;   // 4096 tokens

// ---- helpers -------------------------------------------------------------

DEV unsigned short f2bf(float f) {           // f32 -> bf16, round-nearest-even
  unsigned u = __builtin_bit_cast(unsigned, f);
  u += 0x7fffu + ((u >> 16) & 1u);
  return (unsigned short)(u >> 16);
}
DEV unsigned pack2bf(float a, float b) {     // two bf16 packed into a dword
  return (unsigned)f2bf(a) | ((unsigned)f2bf(b) << 16);
}

DEV v16bf mkfrag(v8u lo, v8u hi) {
  v16u v = __builtin_shufflevector(lo, hi, 0,1,2,3,4,5,6,7,8,9,10,11,12,13,14,15);
  return __builtin_bit_cast(v16bf, v);
}
DEV v16bf asbf(v16u v) { return __builtin_bit_cast(v16bf, v); }

DEV v8f wmma_bf16(v16bf a, v16bf b, v8f c) {
  return __builtin_amdgcn_wmma_f32_16x16x32_bf16(
      false, a, false, b, (short)0, c, false, false);
}

// TDM: async DMA a [32 rows x 64 cols] bf16 tile (row stride 64) into LDS.
// D# per CDNA5 ISA. Inline asm (portable across
// the 5-arg / 6-arg builtin toolchains). Tracked by TENSORcnt.
DEV void tdm_load_v_tile(unsigned lds_off, unsigned long long gaddr) {
  unsigned w1 = (unsigned)__builtin_amdgcn_readfirstlane((int)lds_off);
  unsigned w2 = (unsigned)__builtin_amdgcn_readfirstlane((int)(unsigned)gaddr);
  unsigned w3 = ((unsigned)__builtin_amdgcn_readfirstlane((int)(unsigned)(gaddr >> 32)) & 0x01ffffffu)
              | (2u << 30);                       // type=2 ("image")
  u32x4 g0 = { 1u /*count=1*/, w1, w2, w3 };
  u32x8 g1 = { 0x00010000u,    // data_size=1 (2 bytes)
               64u << 16,      // tensor_dim0 = 64  (bits 79:48, low half)
               2048u << 16,    // tensor_dim1 = 2048 (bits 111:80, low half)
               64u << 16,      // tile_dim0 = 64    (bits 127:112)
               32u,            // tile_dim1 = 32    (bits 143:128)
               64u,            // tensor_dim0_stride = 64 (bits 207:160)
               0u, 0u };
  asm volatile("tensor_load_to_lds %0, %1" :: "s"(g0), "s"(g1) : "memory");
}

// ---- projection GEMM: C[m,n] = sum_k A[m,k]*W[n,k] + bias[n] -------------
// MODE 0: out bf16, layout Q  [b,h,s,d]
// MODE 1: out bf16, layout Kt [b,h,d,s]   (transposed, packed 16B stores)
// MODE 2: out bf16, layout V  [b,h,s,d]
// MODE 3: A is bf16 context, out f32 row-major [m,n], non-temporal
template <int MODE>
__global__ void mha_proj_gemm(const float* __restrict__ Af,
                              const unsigned short* __restrict__ Ab,
                              const float* __restrict__ W,
                              const float* __restrict__ bias,
                              unsigned short* __restrict__ outB,
                              float* __restrict__ outF) {
  __shared__ unsigned short lx[32][32];   // A tile  [m][k]  bf16
  __shared__ unsigned short lw[32][64];   // W tile  [k][n]  bf16 (pre-transposed)
  __shared__ float tT[8][16][17];         // per-wave epilogue transpose tile

  const int t    = threadIdx.x;
  const int lane = t & 31;
  const int wv   = t >> 5;                  // 8 waves
  const int m0   = blockIdx.y * 32;
  const int n0   = blockIdx.x * 64;
  const int moff = (wv >> 2) * 16;          // 2x4 wave grid -> 32x64 block tile
  const int noff = (wv & 3) * 16;

  // staging coordinates
  const int sm  = t >> 3;            // A: row 0..31, 4 consecutive k
  const int sk  = (t & 7) * 4;
  const int wn  = (t & 31) * 2;      // W: pair of n rows
  const int wkq = (t >> 5) * 4;      // W: 4 consecutive k

  const float*          arowF = (MODE == 3) ? nullptr : &Af[(m0 + sm) * DM + sk];
  const unsigned short* arowB = (MODE == 3) ? &Ab[(m0 + sm) * DM + sk] : nullptr;
  const float* wrow0 = &W[(n0 + wn) * DM + wkq];
  const float* wrow1 = wrow0 + DM;

  v8f acc = {};

  for (int k0 = 0; k0 < DM; k0 += 32) {
    __syncthreads();
    // ---- stage A tile: one packed 8-byte LDS store ----
    if (MODE == 3) {
      *(v4u*)&lx[sm][sk] = *(const v4u*)&arowB[k0];
    } else {
      vf4 a4 = *(const vf4*)&arowF[k0];
      v4u p = { f2bf(a4.x), f2bf(a4.y), f2bf(a4.z), f2bf(a4.w) };
      *(v4u*)&lx[sm][sk] = p;
    }
    // ---- stage W transposed: lw[k][n] = W[n][k]; pack 2 n per b32 store ----
    {
      vf4 w0 = *(const vf4*)&wrow0[k0];
      vf4 w1 = *(const vf4*)&wrow1[k0];
      *(unsigned*)&lw[wkq + 0][wn] = pack2bf(w0.x, w1.x);
      *(unsigned*)&lw[wkq + 1][wn] = pack2bf(w0.y, w1.y);
      *(unsigned*)&lw[wkq + 2][wn] = pack2bf(w0.z, w1.z);
      *(unsigned*)&lw[wkq + 3][wn] = pack2bf(w0.w, w1.w);
      if (k0 + 32 < DM) {
        __builtin_prefetch(&wrow0[k0 + 32], 0, 3);   // global_prefetch_b8
        __builtin_prefetch(&wrow1[k0 + 32], 0, 3);
      }
    }
    __syncthreads();

    // ---- one 16x16x32 WMMA step ----
    const int arow = moff + (lane & 15);
    const int abse = (lane < 16) ? 0 : 8;
    v8u alo = *(const v8u*)&lx[arow][abse];
    v8u ahi = *(const v8u*)&lx[arow][abse + 16];
    const int bn = noff + (lane & 15);
    v16u bvv;
    #pragma unroll
    for (int s = 0; s < 16; ++s) bvv[s] = lw[abse + s + (s & 8)][bn];
    acc = wmma_bf16(mkfrag(alo, ahi), asbf(bvv), acc);
  }

  // ---- epilogue: bias, transpose via wave-private LDS, packed stores ----
  const int nn    = n0 + noff + (lane & 15);
  const float bvl = bias[nn];
  const int chalf = (lane < 16) ? 0 : 8;

  #pragma unroll
  for (int r = 0; r < 8; ++r)
    tT[wv][r + chalf][lane & 15] = acc[r] + bvl;     // wave-private, in-order DS

  const int half = lane >> 4;

  if constexpr (MODE == 1) {
    // Kt [b,h,d,s]: one 16B bf16 store along s per lane
    const int N  = lane & 15;
    const int n  = n0 + noff + N;
    const int h  = n >> 6, dd = n & 63;
    const int m  = m0 + moff;
    const int b  = m >> 11, s0 = m & (SEQ - 1);
    v8u p;
    #pragma unroll
    for (int j = 0; j < 8; ++j) p[j] = f2bf(tT[wv][half * 8 + j][N]);
    *(v8u*)&outB[(((b * NH + h) * DKH + dd) << 11) + s0 + half * 8] = p;
  } else if constexpr (MODE == 3) {
    // out f32 row-major: two NT vector stores per lane
    const int sL = lane & 15;
    const int m  = m0 + moff + sL;
    const int nb = n0 + noff + half * 8;
    vf4 f0 = { tT[wv][sL][half * 8 + 0], tT[wv][sL][half * 8 + 1],
               tT[wv][sL][half * 8 + 2], tT[wv][sL][half * 8 + 3] };
    vf4 f1 = { tT[wv][sL][half * 8 + 4], tT[wv][sL][half * 8 + 5],
               tT[wv][sL][half * 8 + 6], tT[wv][sL][half * 8 + 7] };
    __builtin_nontemporal_store(f0, (vf4*)&outF[m * DM + nb]);
    __builtin_nontemporal_store(f1, (vf4*)&outF[m * DM + nb + 4]);
  } else {
    // Q/V [b,h,s,d]: one 16B bf16 store along d per lane
    const int sL = lane & 15;
    const int m  = m0 + moff + sL;
    const int b  = m >> 11, s = m & (SEQ - 1);
    const int nb = n0 + noff + half * 8;
    const int h  = nb >> 6, dd0 = nb & 63;
    v8u p;
    #pragma unroll
    for (int j = 0; j < 8; ++j) p[j] = f2bf(tT[wv][sL][half * 8 + j]);
    *(v8u*)&outB[((((b * NH + h) << 11) + s) << 6) + dd0] = p;
  }
}

// ---- fused attention: scores -> softmax (+p_attn) -> P@V (TDM-fed) ------
__global__ void mha_attn(const unsigned short* __restrict__ Q,   // [b,h,s,d] bf16
                         const unsigned short* __restrict__ Kt,  // [b,h,d,s] bf16
                         const unsigned short* __restrict__ V,   // [b,h,s,d] bf16
                         const int* __restrict__ mask,           // [b,s,s]
                         float* __restrict__ Pout,               // [b,h,s,s] f32
                         unsigned short* __restrict__ Xc) {      // [b,s,1024] bf16
  extern __shared__ char smem_raw[];
  float* sc   = (float*)smem_raw;              // [16][SEQ]      128 KB
  float* pacc = sc + 16 * SEQ;                 // [8][4][16][16]  32 KB
  float* red  = pacc + 8 * 4 * 256;            // [16][16]         1 KB
  unsigned short* vstage = (unsigned short*)(red + 256);  // 8 waves x 2 x 32x64 bf16 = 64 KB

  const int t    = threadIdx.x;
  const int lane = t & 31;
  const int wv   = t >> 5;
  const int qt   = blockIdx.x, h = blockIdx.y, b = blockIdx.z;
  const int q0   = qt * 16;
  const size_t bh = (size_t)(b * NH + h);

  const unsigned short* Qb  = Q  + (bh * SEQ + q0) * DKH;
  const unsigned short* Ktb = Kt + bh * DKH * SEQ;
  const unsigned short* Vb  = V  + bh * SEQ * DKH;

  const int chalf = (lane < 16) ? 0 : 8;
  const int abse  = chalf;                     // A-frag K-chunk base (0 or 8)

  // ---- Q fragments (two A frags: d=0..31, d=32..63) --------------------
  const int qrow = lane & 15;
  v16bf qa0, qa1;
  {
    v8u a = *(const v8u*)&Qb[qrow * DKH + abse];
    v8u c = *(const v8u*)&Qb[qrow * DKH + abse + 16];
    qa0 = mkfrag(a, c);
    v8u d = *(const v8u*)&Qb[qrow * DKH + 32 + abse];
    v8u e = *(const v8u*)&Qb[qrow * DKH + 32 + abse + 16];
    qa1 = mkfrag(d, e);
  }

  // ---- phase 1: scores = (Q K^T)/8 into LDS ----------------------------
  const int bk = (lane & 15) + ((lane < 16) ? 0 : 16);
  for (int kt = wv; kt < SEQ / 16; kt += 8) {
    const int key0 = kt * 16;
    v16u b0, b1;
    #pragma unroll
    for (int s = 0; s < 16; ++s) {
      const int d = abse + s + (s & 8);
      b0[s] = Ktb[(size_t)d * SEQ + key0 + (lane & 15)];
      b1[s] = Ktb[(size_t)(32 + d) * SEQ + key0 + (lane & 15)];
    }
    v8f a = {};
    a = wmma_bf16(qa0, asbf(b0), a);
    a = wmma_bf16(qa1, asbf(b1), a);
    #pragma unroll
    for (int r = 0; r < 8; ++r)
      sc[(r + chalf) * SEQ + key0 + (lane & 15)] = a[r] * 0.125f;
  }
  __syncthreads();

  // ---- phase 2: masked softmax (vectorized), write p_attn (NT) ---------
  {
    const int r  = t >> 4;          // row 0..15
    const int c0 = t & 15;          // 16 threads per row, 4-wide chunks
    const vi4*   mrow4 = (const vi4*)(mask + ((size_t)b * SEQ + q0 + r) * SEQ);
    vf4*         srow4 = (vf4*)(sc + r * SEQ);
    vf4*         prow4 = (vf4*)(Pout + (bh * SEQ + q0 + r) * SEQ);

    float lm = -3.0e38f;
    #pragma unroll 4
    for (int j = 0; j < SEQ / 64; ++j) {
      int idx = c0 + 16 * j;
      vf4 s = srow4[idx];
      vi4 m = __builtin_nontemporal_load(&mrow4[idx]);
      if (m.x == 0) s.x = -1.0e9f;
      if (m.y == 0) s.y = -1.0e9f;
      if (m.z == 0) s.z = -1.0e9f;
      if (m.w == 0) s.w = -1.0e9f;
      srow4[idx] = s;
      lm = fmaxf(lm, fmaxf(fmaxf(s.x, s.y), fmaxf(s.z, s.w)));
    }
    red[r * 16 + c0] = lm;
    __syncthreads();
    float rm = -3.0e38f;
    for (int i = 0; i < 16; ++i) rm = fmaxf(rm, red[r * 16 + i]);
    __syncthreads();
    float ls = 0.f;
    #pragma unroll 4
    for (int j = 0; j < SEQ / 64; ++j) {
      int idx = c0 + 16 * j;
      vf4 s = srow4[idx];
      vf4 e = { __expf(s.x - rm), __expf(s.y - rm),
                __expf(s.z - rm), __expf(s.w - rm) };
      srow4[idx] = e;
      ls += (e.x + e.y) + (e.z + e.w);
    }
    red[r * 16 + c0] = ls;
    __syncthreads();
    float sum = 0.f;
    for (int i = 0; i < 16; ++i) sum += red[r * 16 + i];
    float inv = 1.0f / sum;
    #pragma unroll 4
    for (int j = 0; j < SEQ / 64; ++j) {
      int idx = c0 + 16 * j;
      vf4 e = srow4[idx];
      vf4 p = { e.x * inv, e.y * inv, e.z * inv, e.w * inv };
      srow4[idx] = p;
      __builtin_nontemporal_store(p, &prow4[idx]);
    }
  }
  __syncthreads();

  // ---- phase 3: X = P @ V; each wave owns 256 keys, all 4 d-tiles ------
  // V tiles are DMA'd by the Tensor Data Mover, double-buffered.
  unsigned short* vbuf0 = vstage + wv * 4096;           // [32][64] bf16
  unsigned short* vbuf1 = vbuf0 + 2048;
  const unsigned lds0 = (unsigned)(unsigned long long)(void*)vbuf0;
  const unsigned lds1 = (unsigned)(unsigned long long)(void*)vbuf1;
  const unsigned long long vbase = (unsigned long long)(const void*)Vb;
  const int kw0 = wv * 256;                              // this wave's key base

  tdm_load_v_tile(lds0, vbase + (unsigned long long)kw0 * DKH * 2);

  v8f acc0 = {}, acc1 = {}, acc2 = {}, acc3 = {};
  const int brow = (lane & 15) + ((lane < 16) ? 0 : 16);  // key row in V tile

  for (int st = 0; st < 8; ++st) {
    if (st + 1 < 8)
      tdm_load_v_tile((st & 1) ? lds0 : lds1,
                      vbase + (unsigned long long)(kw0 + (st + 1) * 32) * DKH * 2);
    // A frag (P, f32 -> bf16) built once from 4x ds_load_b128; reused x4
    const int k0 = kw0 + st * 32;
    const float* srowA = sc + (lane & 15) * SEQ + k0 + abse;
    vf4 f0 = ((const vf4*)srowA)[0];
    vf4 f1 = ((const vf4*)srowA)[1];
    vf4 f2 = ((const vf4*)(srowA + 16))[0];
    vf4 f3 = ((const vf4*)(srowA + 16))[1];
    v16u av = { f2bf(f0.x), f2bf(f0.y), f2bf(f0.z), f2bf(f0.w),
                f2bf(f1.x), f2bf(f1.y), f2bf(f1.z), f2bf(f1.w),
                f2bf(f2.x), f2bf(f2.y), f2bf(f2.z), f2bf(f2.w),
                f2bf(f3.x), f2bf(f3.y), f2bf(f3.z), f2bf(f3.w) };
    v16bf af = asbf(av);

    // s_wait_tensorcnt needs a constant immediate; branch is wave-uniform
    if (st + 1 < 8) __builtin_amdgcn_s_wait_tensorcnt(1);
    else            __builtin_amdgcn_s_wait_tensorcnt(0);

    const unsigned short* vb = (st & 1) ? vbuf1 : vbuf0;
    v16u v0, v1, v2, v3;
    #pragma unroll
    for (int s = 0; s < 16; ++s) {
      const unsigned short* vr = &vb[(abse + s + (s & 8)) * DKH + (lane & 15)];
      v0[s] = vr[0]; v1[s] = vr[16]; v2[s] = vr[32]; v3[s] = vr[48];
    }
    acc0 = wmma_bf16(af, asbf(v0), acc0);
    acc1 = wmma_bf16(af, asbf(v1), acc1);
    acc2 = wmma_bf16(af, asbf(v2), acc2);
    acc3 = wmma_bf16(af, asbf(v3), acc3);
  }

  // partials -> LDS
  {
    float* pw = pacc + wv * 4 * 256;
    #pragma unroll
    for (int r = 0; r < 8; ++r) {
      int idx = (r + chalf) * 16 + (lane & 15);
      pw[0 * 256 + idx] = acc0[r];
      pw[1 * 256 + idx] = acc1[r];
      pw[2 * 256 + idx] = acc2[r];
      pw[3 * 256 + idx] = acc3[r];
    }
  }
  __syncthreads();

  // waves 0..3 reduce the 8 partials of their d-tile and store context
  if (wv < 4) {
    const int nt = wv;
    #pragma unroll
    for (int r = 0; r < 8; ++r) {
      int M = r + chalf, N = lane & 15;
      int idx = (nt * 16 + M) * 16 + N;
      float s = 0.f;
      #pragma unroll
      for (int w = 0; w < 8; ++w) s += pacc[w * 4 * 256 + idx];
      Xc[((size_t)(b * SEQ + q0 + M)) * DM + h * DKH + nt * 16 + N] = f2bf(s);
    }
  }
}

// ---- host-side launcher --------------------------------------------------

extern "C" void kernel_launch(void* const* d_in, const int* in_sizes, int n_in,
                              void* d_out, int out_size, void* d_ws, size_t ws_size,
                              hipStream_t stream) {
  (void)in_sizes; (void)n_in; (void)out_size; (void)ws_size;
  const float* query = (const float*)d_in[0];
  const float* key   = (const float*)d_in[1];
  const float* value = (const float*)d_in[2];
  const int*   mask  = (const int*)d_in[3];
  const float* Wq = (const float*)d_in[4];  const float* bq = (const float*)d_in[5];
  const float* Wk = (const float*)d_in[6];  const float* bk = (const float*)d_in[7];
  const float* Wv = (const float*)d_in[8];  const float* bv = (const float*)d_in[9];
  const float* Wo = (const float*)d_in[10]; const float* bo = (const float*)d_in[11];

  const size_t perTensor = (size_t)BATCH * NH * SEQ * DKH;   // 4,194,304 elements
  unsigned short* q_bf  = (unsigned short*)d_ws;
  unsigned short* kt_bf = q_bf  + perTensor;
  unsigned short* v_bf  = kt_bf + perTensor;
  unsigned short* x_bf  = v_bf  + perTensor;                 // context, bf16

  float* out    = (float*)d_out;                             // [B,S,DM]
  float* p_attn = out + (size_t)BATCH * SEQ * DM;            // [B,H,S,S]

  dim3 gblk(DM / 64, ROWS / 32);   // 16 x 128
  dim3 tblk(256);

  mha_proj_gemm<0><<<gblk, tblk, 0, stream>>>(query, nullptr, Wq, bq, q_bf,  nullptr);
  mha_proj_gemm<1><<<gblk, tblk, 0, stream>>>(key,   nullptr, Wk, bk, kt_bf, nullptr);
  mha_proj_gemm<2><<<gblk, tblk, 0, stream>>>(value, nullptr, Wv, bv, v_bf,  nullptr);

  // 128K (scores) + 32K (partials) + 1K (red) + 64K (V staging) = 225 KB
  const size_t smem = (size_t)(16 * SEQ + 8 * 4 * 256 + 256) * sizeof(float)
                    + (size_t)8 * 2 * 32 * DKH * sizeof(unsigned short);
  dim3 gattn(SEQ / 16, NH, BATCH);
  mha_attn<<<gattn, tblk, smem, stream>>>(q_bf, kt_bf, v_bf, mask, p_attn, x_bf);

  mha_proj_gemm<3><<<gblk, tblk, 0, stream>>>(nullptr, x_bf, Wo, bo, nullptr, out);
}
